// MatrixLayer_36610301231304
// MI455X (gfx1250) — matrix operand order repaired and checked by reading, not compiled, not run on hardware
//
#include <hip/hip_runtime.h>

// CDNA5 / gfx1250 implicit-GEMM 3x3 conv via v_wmma_f32_16x16x32_bf16.
// GEMM view: D[co, px] = sum_k W^T[co, k] * Im2col[k, px], K iterated as
// (tap position outer, input channel inner). Branch-free halo handling,
// packed bf16 LDS stores, single-base-pointer loads with immediate offsets.

typedef __attribute__((ext_vector_type(16))) __bf16 v16bf;
typedef __attribute__((ext_vector_type(8)))  __bf16 v8bf;
typedef __attribute__((ext_vector_type(8)))  float  v8f;

#define IMG_H   56
#define IMG_W   56
#define IMG_HW  3136          // 56*56
#define C_IN    256
#define C_OUT   256
#define NBATCH  32
#define BPX     128           // pixels per block  (GEMM N)
#define BCO     128           // out-chans per block (GEMM M)
#define BK      32            // K chunk (one WMMA K)
#define PITCH   40            // LDS row pitch in bf16 (80B) -> conflict-free b128 reads
#define WROW    (9 * C_OUT)   // weight row stride per input channel (2304 floats)

union PkBf16 {
    __bf16   h[2];
    unsigned u;
};

__global__ __launch_bounds__(256)
void conv3x3_wmma_bf16(const float* __restrict__ x,
                       const float* __restrict__ w,
                       float* __restrict__ out)
{
    __shared__ __bf16 Ws[BCO * PITCH];  // [co][k]  (weights, transposed)
    __shared__ __bf16 Xs[BPX * PITCH];  // [px][k]  (im2col slice)

    const int tid      = threadIdx.x;
    const int lane     = tid & 31;
    const int wv       = tid >> 5;       // 0..7
    const int wm       = wv & 1;         // co half   (64 rows)
    const int wn       = wv >> 1;        // px quarter (32 cols)
    const int laneIdx  = lane & 15;
    const int laneHalf = lane >> 4;

    const int pxBlock = (blockIdx.x >> 1) * BPX;   // 784 pixel tiles
    const int coBlock = (blockIdx.x & 1)  * BCO;   // 2 channel tiles

    // cooperative loader mapping: 128 rows x 2 contiguous 16-channel halves
    const int l_row = tid & 127;         // px for X loads, co for W loads
    const int l_ch  = (tid >> 7) * 16;   // 0 or 16: this thread's channel base

    // decompose this thread's loader pixel once
    const int px   = pxBlock + l_row;
    const int pn   = px / IMG_HW;
    const int prem = px - pn * IMG_HW;
    const int py   = prem / IMG_W;
    const int pxx  = prem - py * IMG_W;

    v8f acc[4][2] = {};   // [co tile][px tile], 8 f32 VGPRs each

    for (int pos = 0; pos < 9; ++pos) {
        const int dy = pos / 3 - 1;
        const int dx = pos - (pos / 3) * 3 - 1;
        const int iy = py + dy;
        const int ix = pxx + dx;
        const bool valid = (iy >= 0) & (iy < IMG_H) & (ix >= 0) & (ix < IMG_W);
        // clamp into bounds so loads are always legal; zero packed value below
        const int iyc = iy < 0 ? 0 : (iy >= IMG_H ? IMG_H - 1 : iy);
        const int ixc = ix < 0 ? 0 : (ix >= IMG_W ? IMG_W - 1 : ix);

        // single base pointers; all unrolled loads use constant immediate offsets
        const float* xp = x + (long)pn * (C_IN * IMG_HW) + (long)iyc * IMG_W + ixc
                            + (long)l_ch * IMG_HW;
        const float* wp = w + (long)(l_ch * 9 + pos) * C_OUT + coBlock + l_row;

        for (int ck = 0; ck < C_IN; ck += BK) {
            // ---- stage X tile (im2col), fp32 -> packed bf16 ----
#pragma unroll
            for (int i = 0; i < 8; ++i) {
                const int c = l_ch + 2 * i;
                const float f0 = xp[(2 * i)     * IMG_HW];
                const float f1 = xp[(2 * i + 1) * IMG_HW];
                PkBf16 p;
                p.h[0] = (__bf16)f0;
                p.h[1] = (__bf16)f1;
                const unsigned v = valid ? p.u : 0u;   // one v_cndmask per pair
                *reinterpret_cast<unsigned*>(&Xs[l_row * PITCH + c]) = v;
            }
            // ---- stage W tile transposed: Ws[co][c] = w[(c*9+pos)][co] ----
#pragma unroll
            for (int i = 0; i < 8; ++i) {
                const int c = l_ch + 2 * i;
                const float f0 = wp[(2 * i)     * WROW];
                const float f1 = wp[(2 * i + 1) * WROW];
                PkBf16 p;
                p.h[0] = (__bf16)f0;
                p.h[1] = (__bf16)f1;
                *reinterpret_cast<unsigned*>(&Ws[l_row * PITCH + c]) = p.u;
            }
            // prefetch next channel chunk of x (address always valid: clamped)
            if ((ck + BK) < C_IN)
                __builtin_prefetch(xp + (long)BK * IMG_HW, 0, 1);

            xp += (long)BK * IMG_HW;   // one u64 add per chunk
            wp += (long)BK * WROW;     // one u64 add per chunk

            __syncthreads();

            // ---- per-wave fragments ----
            // A (16x32 bf16): lanes 0-15 hold K 0-7 & 16-23, lanes 16-31 hold K 8-15 & 24-31
            v16bf afr[4];
#pragma unroll
            for (int mt = 0; mt < 4; ++mt) {
                const int row = wm * 64 + mt * 16 + laneIdx;
                const __bf16* p = &Ws[row * PITCH + laneHalf * 8];
                *(v8bf*)&afr[mt]       = *(const v8bf*)p;
                *((v8bf*)&afr[mt] + 1) = *(const v8bf*)(p + 16);
            }
            // B (32x16 bf16): lanes 0-15 hold K 0-7 & 16-23, lanes 16-31 hold K 8-15 & 24-31
            v16bf bfr[2];
#pragma unroll
            for (int nt = 0; nt < 2; ++nt) {
                const int row = wn * 32 + nt * 16 + laneIdx;
                const __bf16* p = &Xs[row * PITCH + laneHalf * 8];
                *(v8bf*)&bfr[nt]       = *(const v8bf*)p;
                *((v8bf*)&bfr[nt] + 1) = *(const v8bf*)(p + 16);
            }

            // ---- 8 WMMAs per K-chunk per wave ----
#pragma unroll
            for (int mt = 0; mt < 4; ++mt)
#pragma unroll
                for (int nt = 0; nt < 2; ++nt)
                    acc[mt][nt] = __builtin_amdgcn_wmma_f32_16x16x32_bf16(
                        false, afr[mt], false, bfr[nt],
                        (short)0, acc[mt][nt], false, false);

            __syncthreads();
        }
    }

    // ---- epilogue: D rows = co, cols(lanes) = pixels -> coalesced 64B runs ----
#pragma unroll
    for (int nt = 0; nt < 2; ++nt) {
        const int opx  = pxBlock + wn * 32 + nt * 16 + laneIdx;
        const int on   = opx / IMG_HW;
        const int orem = opx - on * IMG_HW;
        // single base per nt; all stores use constant immediate offsets
        float* op = out + (long)on * (C_OUT * IMG_HW) + orem
                        + (long)(coBlock + wm * 64 + laneHalf * 8) * IMG_HW;
#pragma unroll
        for (int mt = 0; mt < 4; ++mt)
#pragma unroll
            for (int r = 0; r < 8; ++r)
                op[(mt * 16 + r) * IMG_HW] = acc[mt][nt][r];
    }
}

extern "C" void kernel_launch(void* const* d_in, const int* in_sizes, int n_in,
                              void* d_out, int out_size, void* d_ws, size_t ws_size,
                              hipStream_t stream) {
    (void)in_sizes; (void)n_in; (void)d_ws; (void)ws_size; (void)out_size;
    const float* x = (const float*)d_in[0];            // [32, 256, 56, 56] f32
    const float* w = (const float*)d_in[1];            // [2304, 256] f32
    float* out     = (float*)d_out;                    // [32, 256, 56, 56] f32

    const int pxTiles = (NBATCH * IMG_HW) / BPX;       // 784
    const int coTiles = C_OUT / BCO;                   // 2
    dim3 grid(pxTiles * coTiles), block(256);
    conv3x3_wmma_bf16<<<grid, block, 0, stream>>>(x, w, out);
}
